// SparseTransformer_48146583388632
// MI455X (gfx1250) — compile-verified, matrix operand order checked by reading, not run on hardware
//
#include <hip/hip_runtime.h>

// ---------------- static config (matches reference) ----------------
#define BATCH 8
#define SEQ   784
#define DIM   512
#define HEADS 8
#define DHD   64
#define TOK   (BATCH*SEQ)      // 6272
#define WINW  196
#define CBS   392
#define CST   196
#define SBS   392
#define NCB   3
#define NSB   2
#define FFD   2048
#define VT_LD 800              // padded v^T row (784 -> 800)
#define EPS   1e-5f

typedef __bf16        v16bf __attribute__((ext_vector_type(16)));
typedef float         v8f   __attribute__((ext_vector_type(8)));
typedef unsigned int  u32x4 __attribute__((ext_vector_type(4)));
typedef int           i32x4 __attribute__((ext_vector_type(4)));
typedef int           i32x8 __attribute__((ext_vector_type(8)));

#if defined(__has_builtin)
#if __has_builtin(__builtin_amdgcn_tensor_load_to_lds) && __has_builtin(__builtin_amdgcn_s_wait_tensorcnt)
#define HAVE_TDM 1
#endif
#endif

union UFrag { unsigned int u[8]; v16bf v; };

__device__ inline unsigned short f2bf(float f) {
    unsigned int u = __float_as_uint(f);
    u += 0x7FFFu + ((u >> 16) & 1u);          // round-to-nearest-even
    return (unsigned short)(u >> 16);
}
__device__ inline float bf2f(unsigned short u) {
    return __uint_as_float(((unsigned int)u) << 16);
}
__device__ inline v8f wmma_bf16(const v16bf& a, const v16bf& b, const v8f& c) {
    return __builtin_amdgcn_wmma_f32_16x16x32_bf16(false, a, false, b, (short)0, c, false, false);
}
__device__ inline v8f zero8() {
    v8f z = {0.f,0.f,0.f,0.f,0.f,0.f,0.f,0.f};
    return z;
}

#ifdef HAVE_TDM
// Issue one TDM 2-D tile load: gptr -> LDS[lds_off], tile = tile1 rows x tile0 elems (bf16),
// global row stride = stride0 elements. D# layout per CDNA5 ISA ch.10.8 (groups 0/1; 2/3 zero).
__device__ inline void tdm_load_tile_bf16(const unsigned short* gptr, unsigned lds_off,
                                          unsigned tdim0, unsigned tdim1,
                                          unsigned tile0, unsigned tile1, unsigned stride0) {
    unsigned long long ga = (unsigned long long)gptr;
    u32x4 g0;
    g0[0] = 1u;                                             // count=1, user desc
    g0[1] = lds_off;                                        // lds_addr (bytes)
    g0[2] = (unsigned)(ga & 0xFFFFFFFFull);                 // global_addr[31:0]
    g0[3] = (unsigned)((ga >> 32) & 0x1FFFFFFull) | (2u << 30);  // addr[56:32] | type=2
    i32x8 g1;
    g1[0] = (int)(1u << 16);                                // data_size=1 (2 bytes)
    g1[1] = (int)((tdim0 & 0xFFFFu) << 16);                 // tensor_dim0[15:0]
    g1[2] = (int)((tdim0 >> 16) | ((tdim1 & 0xFFFFu) << 16));
    g1[3] = (int)((tdim1 >> 16) | (tile0 << 16));           // tile_dim0
    g1[4] = (int)(tile1);                                   // tile_dim1 (tile_dim2=0)
    g1[5] = (int)stride0;                                   // tensor_dim0_stride[31:0]
    g1[6] = 0;
    g1[7] = 0;
    i32x4 z4 = {0, 0, 0, 0};
#if defined(__clang_major__) && __clang_major__ >= 23
    i32x8 z8 = {0, 0, 0, 0, 0, 0, 0, 0};
    __builtin_amdgcn_tensor_load_to_lds(g0, g1, z4, z4, z8, 0);
#else
    __builtin_amdgcn_tensor_load_to_lds(g0, g1, z4, z4, 0);
#endif
}
#endif

// ---------------- layout / pointwise kernels ----------------

__global__ void posadd_k(const float* __restrict__ x, const float* __restrict__ pos,
                         float* __restrict__ xs) {
    int idx = blockIdx.x * blockDim.x + threadIdx.x;
    if (idx >= TOK * DIM) return;
    int c = idx % DIM, t = idx / DIM;
    int b = t / SEQ,  n = t % SEQ;
    xs[idx] = x[((size_t)(b * DIM + c)) * SEQ + n] + pos[n];
}

__global__ void outtr_k(const float* __restrict__ xs, float* __restrict__ out) {
    int idx = blockIdx.x * blockDim.x + threadIdx.x;
    if (idx >= BATCH * DIM * SEQ) return;
    int n = idx % SEQ, c = (idx / SEQ) % DIM, b = idx / (SEQ * DIM);
    out[idx] = xs[((size_t)(b * SEQ + n)) * DIM + c];
}

// W[K,N] fp32 -> Wt[N,K] bf16
__global__ void convwt_k(const float* __restrict__ W, unsigned short* __restrict__ Wt,
                         int K, int N) {
    int idx = blockIdx.x * blockDim.x + threadIdx.x;
    if (idx >= K * N) return;
    int k = idx % K, n = idx / K;
    Wt[(size_t)n * K + k] = f2bf(W[(size_t)k * N + n]);
}

__global__ void convw_k(const float* __restrict__ W, unsigned short* __restrict__ Wb, int n) {
    int idx = blockIdx.x * blockDim.x + threadIdx.x;
    if (idx < n) Wb[idx] = f2bf(W[idx]);
}

__global__ __launch_bounds__(256) void rmsnorm_k(const float* __restrict__ xs,
                                                 const float* __restrict__ g,
                                                 unsigned short* __restrict__ out) {
    int t = blockIdx.x;
    __shared__ float red[256];
    float s = 0.f;
    for (int c = threadIdx.x; c < DIM; c += 256) {
        float v = xs[(size_t)t * DIM + c];
        s += v * v;
    }
    red[threadIdx.x] = s; __syncthreads();
    for (int w = 128; w > 0; w >>= 1) {
        if ((int)threadIdx.x < w) red[threadIdx.x] += red[threadIdx.x + w];
        __syncthreads();
    }
    float inv = rsqrtf(red[0] / DIM + EPS);
    for (int c = threadIdx.x; c < DIM; c += 256)
        out[(size_t)t * DIM + c] = f2bf(xs[(size_t)t * DIM + c] * inv * g[c]);
}

__global__ __launch_bounds__(256) void layernorm_k(const float* __restrict__ xs,
                                                   const float* __restrict__ g,
                                                   const float* __restrict__ bb,
                                                   unsigned short* __restrict__ out) {
    int t = blockIdx.x;
    __shared__ float r0[256], r1[256];
    float s = 0.f, s2 = 0.f;
    for (int c = threadIdx.x; c < DIM; c += 256) {
        float v = xs[(size_t)t * DIM + c];
        s += v; s2 += v * v;
    }
    r0[threadIdx.x] = s; r1[threadIdx.x] = s2; __syncthreads();
    for (int w = 128; w > 0; w >>= 1) {
        if ((int)threadIdx.x < w) { r0[threadIdx.x] += r0[threadIdx.x + w]; r1[threadIdx.x] += r1[threadIdx.x + w]; }
        __syncthreads();
    }
    float mean = r0[0] / DIM;
    float var  = r1[0] / DIM - mean * mean;
    float inv  = rsqrtf(var + EPS);
    for (int c = threadIdx.x; c < DIM; c += 256)
        out[(size_t)t * DIM + c] = f2bf((xs[(size_t)t * DIM + c] - mean) * inv * g[c] + bb[c]);
}

__global__ void gates_k(const unsigned short* __restrict__ xr, const float* __restrict__ Wg,
                        float* __restrict__ gates) {
    int idx = blockIdx.x * blockDim.x + threadIdx.x;
    if (idx >= TOK * HEADS * 3) return;
    int j = idx % (HEADS * 3), t = idx / (HEADS * 3);
    float s = 0.f;
    for (int c = 0; c < DIM; c++)
        s += bf2f(xr[(size_t)t * DIM + c]) * Wg[(size_t)c * (HEADS * 3) + j];
    gates[idx] = 1.f / (1.f + __expf(-s));
}

__global__ void pool_k(const float* __restrict__ kf, const float* __restrict__ vf,
                       float* __restrict__ ck, float* __restrict__ cv) {
    int idx = blockIdx.x * blockDim.x + threadIdx.x;
    if (idx >= BATCH * NCB * HEADS * DHD) return;
    int d = idx % DHD, h = (idx / DHD) % HEADS, j = (idx / (DHD * HEADS)) % NCB,
        b = idx / (DHD * HEADS * NCB);
    float sk = 0.f, sv = 0.f;
    for (int n = j * CST; n < j * CST + CBS; n++) {
        size_t o = ((size_t)(b * SEQ + n) * HEADS + h) * DHD + d;
        sk += kf[o]; sv += vf[o];
    }
    size_t oo = ((size_t)(b * NCB + j) * HEADS + h) * DHD + d;
    ck[oo] = sk / CBS; cv[oo] = sv / CBS;
}

__global__ void comp_k(const float* __restrict__ q, const float* __restrict__ ck,
                       const float* __restrict__ cv, float* __restrict__ outc,
                       int* __restrict__ sel) {
    int idx = blockIdx.x * blockDim.x + threadIdx.x;
    if (idx >= BATCH * HEADS * SEQ) return;
    int i = idx % SEQ, h = (idx / SEQ) % HEADS, b = idx / (SEQ * HEADS);
    const float* qp = q + ((size_t)(b * SEQ + i) * HEADS + h) * DHD;
    const float* c0 = ck + ((size_t)(b * NCB + 0) * HEADS + h) * DHD;
    const float* c1 = ck + ((size_t)(b * NCB + 1) * HEADS + h) * DHD;
    const float* c2 = ck + ((size_t)(b * NCB + 2) * HEADS + h) * DHD;
    float s0 = 0.f, s1 = 0.f, s2 = 0.f;
    for (int d = 0; d < DHD; d++) {
        float qd = qp[d];
        s0 += qd * c0[d]; s1 += qd * c1[d]; s2 += qd * c2[d];
    }
    s0 *= 0.125f; s1 *= 0.125f; s2 *= 0.125f;
    float m = fmaxf(s0, fmaxf(s1, s2));
    float e0 = __expf(s0 - m), e1 = __expf(s1 - m), e2 = __expf(s2 - m);
    float inv = 1.f / (e0 + e1 + e2);
    float p0 = e0 * inv, p1 = e1 * inv, p2 = e2 * inv;
    // OVERLAP = [[1,0],[0.5,0.5],[0,1]]
    float imp0 = p0 + 0.5f * p1, imp1 = 0.5f * p1 + p2;
    sel[(size_t)(b * HEADS + h) * SEQ + i] = (imp1 > imp0) ? 1 : 0;
    const float* v0 = cv + ((size_t)(b * NCB + 0) * HEADS + h) * DHD;
    const float* v1 = cv + ((size_t)(b * NCB + 1) * HEADS + h) * DHD;
    const float* v2 = cv + ((size_t)(b * NCB + 2) * HEADS + h) * DHD;
    float* op = outc + ((size_t)(b * SEQ + i) * HEADS + h) * DHD;
    for (int d = 0; d < DHD; d++)
        op[d] = p0 * v0[d] + p1 * v1[d] + p2 * v2[d];
}

__global__ void vtr_k(const float* __restrict__ vf, unsigned short* __restrict__ vt) {
    int idx = blockIdx.x * blockDim.x + threadIdx.x;
    if (idx >= BATCH * HEADS * DHD * VT_LD) return;
    int n = idx % VT_LD, d = (idx / VT_LD) % DHD, h = (idx / (VT_LD * DHD)) % HEADS,
        b = idx / (VT_LD * DHD * HEADS);
    vt[idx] = (n < SEQ) ? f2bf(vf[((size_t)(b * SEQ + n) * HEADS + h) * DHD + d])
                        : (unsigned short)0;
}

// ---------------- WMMA GEMM: C[M,N] = A[M,K] @ Bt[N,K]^T ----------------
// Block = 4 waves, block tile 128(M) x 64(N). The 64x32 bf16 B tile is staged in LDS
// by the Tensor Data Mover (double buffered, TENSORcnt) and shared by all 4 waves.
template<int WF32, int WBF16, int BIAS, int GELU, int RESID>
__global__ __launch_bounds__(128) void gemm_bf16(
    const unsigned short* __restrict__ A, const unsigned short* __restrict__ Bt,
    float* __restrict__ C, unsigned short* __restrict__ Cbf,
    const float* __restrict__ bias, int M, int N, int K) {
    __shared__ unsigned short btile[2][64 * 32];   // 8 KB, double buffered

    int lane = threadIdx.x & 31;
    int wave = threadIdx.x >> 5;
    int r = lane >> 4, nl = lane & 15;
    int m0 = (blockIdx.x * 4 + wave) * 32;
    int n0 = blockIdx.y * 64;

    v8f acc[2][4];
#pragma unroll
    for (int mi = 0; mi < 2; mi++)
#pragma unroll
        for (int ni = 0; ni < 4; ni++) acc[mi][ni] = zero8();

#ifdef HAVE_TDM
    unsigned ldsbase = (unsigned)(unsigned long long)(&btile[0][0]);
    if (threadIdx.x < 32) {      // wave 0 drives the TDM (EXEC ignored, per-wave op)
        tdm_load_tile_bf16(Bt + (size_t)n0 * K, ldsbase, K, N, 32, 64, K);
        __builtin_amdgcn_s_wait_tensorcnt(0);
    }
    __syncthreads();
#endif

    int buf = 0;
    for (int k0 = 0; k0 < K; k0 += 32) {
#ifdef HAVE_TDM
        if (k0 + 32 < K && threadIdx.x < 32)   // async-prefetch next B tile into other buffer
            tdm_load_tile_bf16(Bt + (size_t)n0 * K + k0 + 32, ldsbase + (buf ^ 1) * 4096,
                               K, N, 32, 64, K);
#else
        // cooperative fallback: copy 64x32 bf16 tile (1024 dwords) with 128 threads
        for (int t = threadIdx.x; t < 1024; t += 128) {
            int e = t * 2;
            int row = e >> 5, kk = e & 31;
            ((unsigned int*)&btile[buf][0])[t] =
                *(const unsigned int*)(Bt + (size_t)(n0 + row) * K + k0 + kk);
        }
        __syncthreads();
#endif
        UFrag a[2], bfr[4];
#pragma unroll
        for (int mi = 0; mi < 2; mi++) {
            const unsigned short* ap = A + (size_t)(m0 + mi * 16 + nl) * K + k0 + r * 8;
            if (k0 + 32 < K) __builtin_prefetch(ap + 32, 0, 1);
#pragma unroll
            for (int p = 0; p < 8; p++) {
                int k = ((p >> 2) * 16) + ((p & 3) * 2);
                a[mi].u[p] = *(const unsigned int*)(ap + k);
            }
        }
#pragma unroll
        for (int ni = 0; ni < 4; ni++) {
            const unsigned short* bp = &btile[buf][(ni * 16 + nl) * 32 + r * 8];
#pragma unroll
            for (int p = 0; p < 8; p++) {
                int k = ((p >> 2) * 16) + ((p & 3) * 2);
                bfr[ni].u[p] = *(const unsigned int*)(bp + k);
            }
        }
#pragma unroll
        for (int mi = 0; mi < 2; mi++)
#pragma unroll
            for (int ni = 0; ni < 4; ni++)
                acc[mi][ni] = wmma_bf16(a[mi].v, bfr[ni].v, acc[mi][ni]);
#ifdef HAVE_TDM
        if (threadIdx.x < 32) __builtin_amdgcn_s_wait_tensorcnt(0);
        __syncthreads();
        buf ^= 1;
#else
        __syncthreads();
#endif
    }

#pragma unroll
    for (int mi = 0; mi < 2; mi++)
#pragma unroll
        for (int ni = 0; ni < 4; ni++)
#pragma unroll
            for (int v = 0; v < 8; v++) {
                int row = m0 + mi * 16 + v + 8 * r;
                int col = n0 + ni * 16 + nl;
                float val = acc[mi][ni][v];
                if constexpr (BIAS)  val += bias[col];
                if constexpr (GELU)  val = 0.5f * val * (1.0f + erff(val * 0.70710678118f));
                if constexpr (RESID) val += C[(size_t)row * N + col];
                if constexpr (WF32)  C[(size_t)row * N + col] = val;
                if constexpr (WBF16) Cbf[(size_t)row * N + col] = f2bf(val);
            }
}

// ---------------- fused NSA attention (window + selection branches) ----------------
__global__ __launch_bounds__(32) void attn_fused(
    const unsigned short* __restrict__ qbf, const unsigned short* __restrict__ kbf,
    const unsigned short* __restrict__ vt,  const float* __restrict__ outc,
    const float* __restrict__ gates, const int* __restrict__ sel,
    unsigned short* __restrict__ concat_bf) {
    __shared__ float S[16][VT_LD];
    __shared__ float stats[4][16];   // [wmax, wrsum, smax, srsum][row]

    int lane = threadIdx.x & 31;
    int r = lane >> 4, nl = lane & 15;
    int bt = blockIdx.x, h = blockIdx.y, b = blockIdx.z;
    int i0 = bt * 16;

    // -------- phase 1: score strip S[16,784] = (q @ k^T) * scale --------
    UFrag aq[2];
    {
        const unsigned short* qp = qbf + ((size_t)(b * SEQ + i0 + nl) * HEADS + h) * DHD + r * 8;
#pragma unroll
        for (int ks = 0; ks < 2; ks++)
#pragma unroll
            for (int p = 0; p < 8; p++) {
                int k = ks * 32 + ((p >> 2) * 16) + ((p & 3) * 2);
                aq[ks].u[p] = *(const unsigned int*)(qp + k);
            }
    }
    for (int jt = 0; jt < SEQ / 16; jt++) {
        v8f acc = zero8();
#pragma unroll
        for (int ks = 0; ks < 2; ks++) {
            UFrag bk;
            const unsigned short* kp =
                kbf + ((size_t)(b * SEQ + jt * 16 + nl) * HEADS + h) * DHD + ks * 32 + r * 8;
#pragma unroll
            for (int p = 0; p < 8; p++) {
                int k = ((p >> 2) * 16) + ((p & 3) * 2);
                bk.u[p] = *(const unsigned int*)(kp + k);
            }
            acc = wmma_bf16(aq[ks].v, bk.v, acc);
        }
#pragma unroll
        for (int v = 0; v < 8; v++)
            S[v + 8 * r][jt * 16 + nl] = acc[v] * 0.125f;
    }
    __syncthreads();

    // -------- phase 2: online softmax stats (lanes 0-15 window, 16-31 selection) --------
    {
        int rowm = lane & 15;
        int branch = lane >> 4;
        int i = i0 + rowm;
        int sb = sel[(size_t)(b * HEADS + h) * SEQ + i];
        float mx = -3.0e38f, sum = 0.f;
        for (int j = 0; j < SEQ; j++) {
            int dj = i - j; if (dj < 0) dj = -dj;
            bool ok = (branch == 0) ? (dj < WINW) : ((j / SBS) == sb);
            if (ok) {
                float s = S[rowm][j];
                float mn = fmaxf(mx, s);
                sum = sum * __expf(mx - mn) + __expf(s - mn);
                mx = mn;
            }
        }
        stats[branch * 2 + 0][rowm] = mx;
        stats[branch * 2 + 1][rowm] = 1.f / sum;
    }
    __syncthreads();

    // -------- phase 3: out = P @ V for both branches (P rebuilt in bf16) --------
    float wm = stats[0][nl], wr = stats[1][nl];
    float smx = stats[2][nl], sr = stats[3][nl];
    int im = i0 + nl;
    int sb = sel[(size_t)(b * HEADS + h) * SEQ + im];

    v8f accw[4], accs[4];
#pragma unroll
    for (int ni = 0; ni < 4; ni++) { accw[ni] = zero8(); accs[ni] = zero8(); }

    for (int ks = 0; ks < 25; ks++) {
        int kbase = ks * 32;
        UFrag pw, ps;
#pragma unroll
        for (int p = 0; p < 8; p++) {
            int j0 = kbase + ((p >> 2) * 16) + r * 8 + ((p & 3) * 2);
            unsigned int uw = 0, us = 0;
#pragma unroll
            for (int hh = 0; hh < 2; hh++) {
                int j = j0 + hh;
                float vw = 0.f, vs = 0.f;
                if (j < SEQ) {
                    float s = S[nl][j];
                    int dj = im - j; if (dj < 0) dj = -dj;
                    if (dj < WINW)      vw = __expf(s - wm) * wr;
                    if (j / SBS == sb)  vs = __expf(s - smx) * sr;
                }
                uw |= ((unsigned int)f2bf(vw)) << (16 * hh);
                us |= ((unsigned int)f2bf(vs)) << (16 * hh);
            }
            pw.u[p] = uw; ps.u[p] = us;
        }
        UFrag bv[4];
#pragma unroll
        for (int ni = 0; ni < 4; ni++) {
            const unsigned short* vp =
                vt + ((size_t)((b * HEADS + h) * DHD + ni * 16 + nl)) * VT_LD + kbase + r * 8;
#pragma unroll
            for (int p = 0; p < 8; p++) {
                int k = ((p >> 2) * 16) + ((p & 3) * 2);
                bv[ni].u[p] = *(const unsigned int*)(vp + k);
            }
        }
#pragma unroll
        for (int ni = 0; ni < 4; ni++) {
            accw[ni] = wmma_bf16(pw.v, bv[ni].v, accw[ni]);
            accs[ni] = wmma_bf16(ps.v, bv[ni].v, accs[ni]);
        }
    }

    // -------- phase 4: gated combine with compressed branch --------
#pragma unroll
    for (int ni = 0; ni < 4; ni++)
#pragma unroll
        for (int v = 0; v < 8; v++) {
            int row = v + 8 * r;
            int i = i0 + row;
            int d = ni * 16 + nl;
            size_t gb = (size_t)(b * SEQ + i) * HEADS + h;
            float g0 = gates[gb * 3 + 0], g1 = gates[gb * 3 + 1], g2 = gates[gb * 3 + 2];
            float oc = outc[gb * DHD + d];
            float val = g0 * oc + g1 * accs[ni][v] + g2 * accw[ni][v];
            concat_bf[(size_t)(b * SEQ + i) * DIM + h * DHD + d] = f2bf(val);
        }
}

// ---------------- host side ----------------
extern "C" void kernel_launch(void* const* d_in, const int* in_sizes, int n_in,
                              void* d_out, int out_size, void* d_ws, size_t ws_size,
                              hipStream_t stream) {
    const float* x    = (const float*)d_in[0];
    const float* pos  = (const float*)d_in[1];
    const float* ag   = (const float*)d_in[2];
    const float* Wq   = (const float*)d_in[3];
    const float* Wk   = (const float*)d_in[4];
    const float* Wv   = (const float*)d_in[5];
    const float* Wo   = (const float*)d_in[6];
    const float* Wg   = (const float*)d_in[7];
    const float* ffg  = (const float*)d_in[8];
    const float* ffb  = (const float*)d_in[9];
    const float* w1   = (const float*)d_in[10];
    const float* b1   = (const float*)d_in[11];
    const float* w2   = (const float*)d_in[12];
    const float* b2   = (const float*)d_in[13];

    char* p = (char*)d_ws;
    auto alloc = [&](size_t bytes) -> void* {
        void* r = (void*)p;
        p += (bytes + 255) & ~(size_t)255;
        return r;
    };
    float*          xs     = (float*)alloc((size_t)TOK * DIM * 4);
    unsigned short* xr     = (unsigned short*)alloc((size_t)TOK * DIM * 2);
    float*          qf     = (float*)alloc((size_t)TOK * DIM * 4);
    float*          kf     = (float*)alloc((size_t)TOK * DIM * 4);
    float*          vf     = (float*)alloc((size_t)TOK * DIM * 4);
    unsigned short* qbf    = (unsigned short*)alloc((size_t)TOK * DIM * 2);
    unsigned short* kbf    = (unsigned short*)alloc((size_t)TOK * DIM * 2);
    unsigned short* vt     = (unsigned short*)alloc((size_t)BATCH * HEADS * DHD * VT_LD * 2);
    float*          outc   = (float*)alloc((size_t)TOK * DIM * 4);
    float*          gates  = (float*)alloc((size_t)TOK * HEADS * 3 * 4);
    int*            sel    = (int*)alloc((size_t)BATCH * HEADS * SEQ * 4);
    unsigned short* concat = (unsigned short*)alloc((size_t)TOK * DIM * 2);
    unsigned short* h1     = (unsigned short*)alloc((size_t)TOK * FFD * 2);
    unsigned short* Wqb    = (unsigned short*)alloc((size_t)DIM * DIM * 2);
    unsigned short* Wkb    = (unsigned short*)alloc((size_t)DIM * DIM * 2);
    unsigned short* Wvb    = (unsigned short*)alloc((size_t)DIM * DIM * 2);
    unsigned short* Wob    = (unsigned short*)alloc((size_t)DIM * DIM * 2);
    unsigned short* w1b    = (unsigned short*)alloc((size_t)FFD * DIM * 2);
    unsigned short* w2b    = (unsigned short*)alloc((size_t)DIM * FFD * 2);
    float*          ck     = (float*)alloc((size_t)BATCH * NCB * HEADS * DHD * 4);
    float*          cv     = (float*)alloc((size_t)BATCH * NCB * HEADS * DHD * 4);
    (void)in_sizes; (void)n_in; (void)out_size; (void)ws_size;

    posadd_k<<<(TOK * DIM + 255) / 256, 256, 0, stream>>>(x, pos, xs);

    for (int l = 0; l < 2; l++) {
        const float* Wq_l = Wq + (size_t)l * DIM * DIM;
        const float* Wk_l = Wk + (size_t)l * DIM * DIM;
        const float* Wv_l = Wv + (size_t)l * DIM * DIM;
        const float* Wo_l = Wo + (size_t)l * DIM * DIM;
        const float* Wg_l = Wg + (size_t)l * DIM * HEADS * 3;
        const float* w1_l = w1 + (size_t)l * FFD * DIM;
        const float* w2_l = w2 + (size_t)l * DIM * FFD;

        // --- attention block ---
        rmsnorm_k<<<TOK, 256, 0, stream>>>(xs, ag + (size_t)l * DIM, xr);
        convwt_k<<<(DIM * DIM + 255) / 256, 256, 0, stream>>>(Wq_l, Wqb, DIM, DIM);
        convwt_k<<<(DIM * DIM + 255) / 256, 256, 0, stream>>>(Wk_l, Wkb, DIM, DIM);
        convwt_k<<<(DIM * DIM + 255) / 256, 256, 0, stream>>>(Wv_l, Wvb, DIM, DIM);
        convwt_k<<<(DIM * DIM + 255) / 256, 256, 0, stream>>>(Wo_l, Wob, DIM, DIM);

        gemm_bf16<1, 1, 0, 0, 0><<<dim3(TOK / 128, DIM / 64), 128, 0, stream>>>(
            xr, Wqb, qf, qbf, nullptr, TOK, DIM, DIM);
        gemm_bf16<1, 1, 0, 0, 0><<<dim3(TOK / 128, DIM / 64), 128, 0, stream>>>(
            xr, Wkb, kf, kbf, nullptr, TOK, DIM, DIM);
        gemm_bf16<1, 0, 0, 0, 0><<<dim3(TOK / 128, DIM / 64), 128, 0, stream>>>(
            xr, Wvb, vf, nullptr, nullptr, TOK, DIM, DIM);

        gates_k<<<(TOK * HEADS * 3 + 255) / 256, 256, 0, stream>>>(xr, Wg_l, gates);
        pool_k<<<(BATCH * NCB * HEADS * DHD + 255) / 256, 256, 0, stream>>>(kf, vf, ck, cv);
        comp_k<<<(BATCH * HEADS * SEQ + 255) / 256, 256, 0, stream>>>(qf, ck, cv, outc, sel);
        vtr_k<<<(BATCH * HEADS * DHD * VT_LD + 255) / 256, 256, 0, stream>>>(vf, vt);

        attn_fused<<<dim3(SEQ / 16, HEADS, BATCH), 32, 0, stream>>>(
            qbf, kbf, vt, outc, gates, sel, concat);

        gemm_bf16<1, 0, 0, 0, 1><<<dim3(TOK / 128, DIM / 64), 128, 0, stream>>>(
            concat, Wob, xs, nullptr, nullptr, TOK, DIM, DIM);   // xs += attn_out

        // --- feed-forward block ---
        layernorm_k<<<TOK, 256, 0, stream>>>(xs, ffg + (size_t)l * DIM, ffb + (size_t)l * DIM, xr);
        convw_k<<<(FFD * DIM + 255) / 256, 256, 0, stream>>>(w1_l, w1b, FFD * DIM);
        convw_k<<<(DIM * FFD + 255) / 256, 256, 0, stream>>>(w2_l, w2b, DIM * FFD);

        gemm_bf16<0, 1, 1, 1, 0><<<dim3(TOK / 128, FFD / 64), 128, 0, stream>>>(
            xr, w1b, nullptr, h1, b1 + (size_t)l * FFD, TOK, FFD, DIM);    // GELU(xn@w1^T+b1)
        gemm_bf16<1, 0, 1, 0, 1><<<dim3(TOK / 128, DIM / 64), 128, 0, stream>>>(
            h1, w2b, xs, nullptr, b2 + (size_t)l * DIM, TOK, DIM, FFD);    // xs += h1@w2^T+b2
    }

    outtr_k<<<(BATCH * DIM * SEQ + 255) / 256, 256, 0, stream>>>(xs, (float*)d_out);
}
